// RQ_61916248539278
// MI455X (gfx1250) — matrix operand order repaired and checked by reading, not compiled, not run on hardware
//
#include <hip/hip_runtime.h>

typedef __attribute__((ext_vector_type(16))) __bf16 v16bf;
typedef __attribute__((ext_vector_type(8)))  float  v8f;
typedef __attribute__((ext_vector_type(4)))  int    v4i;

#define N_TOK 65536
#define DIM   256
#define KCB   1024
#define NCODE 4
#define ROWS_PER_BLOCK 128
#define TPB   128                                          // 4 wave32
#define CHUNK 128                                          // codewords per LDS buffer
#define LDS_RES_BYTES (ROWS_PER_BLOCK * DIM * 4)           // 128 KB residual
#define LDS_W2_BYTES  (KCB * 4)                            // 4 KB  0.5*|W|^2
#define LDS_WBUF_BYTES (CHUNK * DIM * 2)                   // 64 KB per W buffer
#define LDS_TOTAL (LDS_RES_BYTES + LDS_W2_BYTES + 2 * LDS_WBUF_BYTES)   // 260 KB

#if defined(__has_builtin)
#if __has_builtin(__builtin_amdgcn_global_load_async_to_lds_b128) && \
    __has_builtin(__builtin_amdgcn_s_wait_asynccnt)
#define HAVE_ASYNC_LDS 1
#endif
#endif
#ifndef HAVE_ASYNC_LDS
#define HAVE_ASYNC_LDS 0
#endif

// Issue a 64 KB codebook chunk copy global->LDS. Async (ASYNCcnt) when the
// gfx1250 builtins exist; plain synchronous copy otherwise.
__device__ __forceinline__ void issue_chunk_copy(const __bf16* __restrict__ gsrc,
                                                 __bf16* __restrict__ ldst,
                                                 int tid) {
#if HAVE_ASYNC_LDS
#pragma unroll
    for (int i = 0; i < (LDS_WBUF_BYTES / 16) / TPB; ++i) {
        const char* gp = (const char*)gsrc + (size_t)(tid + i * TPB) * 16;
        char*       lp = (char*)ldst + (size_t)(tid + i * TPB) * 16;
        __builtin_amdgcn_global_load_async_to_lds_b128(
            (__attribute__((address_space(1))) v4i*)gp,
            (__attribute__((address_space(3))) v4i*)lp, 0, 0);
    }
#else
    const uint4* src = (const uint4*)gsrc;
    uint4*       dst = (uint4*)ldst;
#pragma unroll
    for (int i = 0; i < (LDS_WBUF_BYTES / 16) / TPB; ++i)
        dst[tid + i * TPB] = src[tid + i * TPB];
#endif
}

__device__ __forceinline__ void wait_chunk_copy() {
#if HAVE_ASYNC_LDS
    __builtin_amdgcn_s_wait_asynccnt(0);
#endif
}

// ---------------------------------------------------------------------------
// Prep: codebooks f32 -> bf16, 0.5*|W|^2 per codeword, zero loss slot.
// ---------------------------------------------------------------------------
__global__ void rq_prep(const float* __restrict__ cbf32,
                        __bf16* __restrict__ wbf,
                        float* __restrict__ w2h,
                        float* __restrict__ lossSlot) {
    const int lane = threadIdx.x & 31;
    const int wave = threadIdx.x >> 5;
    const int row  = blockIdx.x * 8 + wave;          // 0 .. 4095
    const float* src = cbf32 + (size_t)row * DIM;
    __bf16*      dst = wbf   + (size_t)row * DIM;
    float ss = 0.f;
#pragma unroll
    for (int j = 0; j < DIM / 32; ++j) {
        float v = src[j * 32 + lane];
        dst[j * 32 + lane] = (__bf16)v;
        ss += v * v;
    }
#pragma unroll
    for (int m = 16; m >= 1; m >>= 1) ss += __shfl_xor(ss, m);
    if (lane == 0) w2h[row] = 0.5f * ss;
    if (blockIdx.x == 0 && threadIdx.x == 0) *lossSlot = 0.f;
}

// ---------------------------------------------------------------------------
// Main RQ kernel. 128 threads (4 wave32) per block, 128 rows per block; each
// wave owns TWO 16-row strips so every B tile fetched from LDS feeds two
// WMMAs. B is double-buffered in VGPRs so the next K-step's ds loads overlap
// the current WMMA pair. Residual f32 in LDS; codebook double-buffered in LDS
// via async global->LDS copies; -0.5*|W|^2 folded into chain-0's C init.
// ---------------------------------------------------------------------------
__global__ void __launch_bounds__(TPB)
rq_main(const float* __restrict__ z,
        const float* __restrict__ cbf32,
        const __bf16* __restrict__ wbf,
        const float* __restrict__ w2h,
        float* __restrict__ outQ,
        int*   __restrict__ outInd,
        float* __restrict__ outLoss) {
    extern __shared__ unsigned char smem[];
    float*  ldsRes = (float*)smem;                                   // 128 KB
    float*  ldsW2  = (float*)(smem + LDS_RES_BYTES);                 // 4 KB
    __bf16* ldsW0  = (__bf16*)(smem + LDS_RES_BYTES + LDS_W2_BYTES); // 64 KB
    __bf16* ldsW1  = (__bf16*)((char*)ldsW0 + LDS_WBUF_BYTES);       // 64 KB

    const int tid  = threadIdx.x;
    const int lane = tid & 31;
    const int wave = tid >> 5;
    const int hf   = (lane >> 4) & 1;      // lane half (WMMA layout)
    const int l16  = lane & 15;
    const size_t rowBase = (size_t)blockIdx.x * ROWS_PER_BLOCK;

    // ---- load z tile into residual LDS ----
    {
        const uint4* src = (const uint4*)(z + rowBase * DIM);
        uint4* dst = (uint4*)ldsRes;
        for (int i = 0; i < (ROWS_PER_BLOCK * DIM / 4) / TPB; ++i)
            dst[tid + i * TPB] = src[tid + i * TPB];
    }
    __syncthreads();

    const int waveRow = wave * 32;                 // this wave's 32-row strip
    float* myRes = ldsRes + waveRow * DIM;

    for (int cbk = 0; cbk < NCODE; ++cbk) {
        // ---- A operands: two 16x256 bf16 strips, per ISA A-layout ----
        // lanes 0-15: M=lane, K = kt*32 + {0..7, 16..23}
        // lanes 16-31: M=lane-16, K = kt*32 + {8..15, 24..31}
        v16bf a0[8], a1[8];
        {
            const float* rrow0 = myRes + l16 * DIM;
            const float* rrow1 = myRes + (16 + l16) * DIM;
#pragma unroll
            for (int kt = 0; kt < 8; ++kt) {
                const int k0 = kt * 32 + hf * 8;
#pragma unroll
                for (int j = 0; j < 8; ++j) {
                    a0[kt][j]     = (__bf16)rrow0[k0 + j];
                    a0[kt][8 + j] = (__bf16)rrow0[k0 + 16 + j];
                    a1[kt][j]     = (__bf16)rrow1[k0 + j];
                    a1[kt][8 + j] = (__bf16)rrow1[k0 + 16 + j];
                }
            }
        }

        float bestV0[8], bestV1[8];
        int   bestI0[8], bestI1[8];
#pragma unroll
        for (int r = 0; r < 8; ++r) {
            bestV0[r] = -3.4e38f; bestI0[r] = 0;
            bestV1[r] = -3.4e38f; bestI1[r] = 0;
        }

        const __bf16* wcb = wbf + (size_t)cbk * KCB * DIM;

        // stage this codebook's 0.5*|W|^2 into LDS
#pragma unroll
        for (int i = 0; i < KCB / TPB; ++i)
            ldsW2[tid + i * TPB] = w2h[cbk * KCB + tid + i * TPB];
        // kick off chunk 0 into buffer 0
        issue_chunk_copy(wcb, ldsW0, tid);
        wait_chunk_copy();
        __syncthreads();

        for (int chunk = 0; chunk < KCB / CHUNK; ++chunk) {
            __bf16* wcur = (chunk & 1) ? ldsW1 : ldsW0;
            __bf16* wnxt = (chunk & 1) ? ldsW0 : ldsW1;
            if (chunk + 1 < KCB / CHUNK) {
                // overlap next chunk's global->LDS copy with this chunk's WMMAs
                issue_chunk_copy(wcb + (size_t)(chunk + 1) * CHUNK * DIM, wnxt, tid);
                __builtin_prefetch(wcb + (size_t)(chunk + 1) * CHUNK * DIM + tid * 256, 0, 1);
            }

            for (int sub = 0; sub < CHUNK / 16; ++sub) {
                const int n = chunk * CHUNK + sub * 16 + l16;   // D-layout column
                const float adj = ldsW2[n];                      // 0.5*|W_n|^2
                // B operand: N = l16, K split across lane halves (ISA B-layout)
                const __bf16* wrow = wcur + (sub * 16 + l16) * DIM + hf * 8;
                v8f c0, c1;
#pragma unroll
                for (int r = 0; r < 8; ++r) c0[r] = -adj;   // bias on chain 0
                c1 = (v8f){0.f, 0.f, 0.f, 0.f, 0.f, 0.f, 0.f, 0.f};

                // software-pipelined B: load next K-step while WMMAs consume
                v16bf b[2];
                *((uint4*)&b[0])     = *(const uint4*)(wrow);
                *((uint4*)&b[0] + 1) = *(const uint4*)(wrow + 16);
#pragma unroll
                for (int kt = 0; kt < 8; ++kt) {
                    if (kt + 1 < 8) {
                        *((uint4*)&b[(kt + 1) & 1])     = *(const uint4*)(wrow + (kt + 1) * 32);
                        *((uint4*)&b[(kt + 1) & 1] + 1) = *(const uint4*)(wrow + (kt + 1) * 32 + 16);
                    }
                    c0 = __builtin_amdgcn_wmma_f32_16x16x32_bf16(
                            false, a0[kt], false, b[kt & 1], (short)0, c0, false, false);
                    c1 = __builtin_amdgcn_wmma_f32_16x16x32_bf16(
                            false, a1[kt], false, b[kt & 1], (short)0, c1, false, false);
                }
#pragma unroll
                for (int r = 0; r < 8; ++r) {
                    float s0 = c0[r];             // dot - 0.5*|W|^2, maximize
                    float s1 = c1[r] - adj;       // de-bias chain 1 here
                    if (s0 > bestV0[r]) { bestV0[r] = s0; bestI0[r] = n; }
                    if (s1 > bestV1[r]) { bestV1[r] = s1; bestI1[r] = n; }
                }
            }
            wait_chunk_copy();
            __syncthreads();
        }

        // ---- argmax across the 16 lanes sharing each M row ----
        // xor masks 1,2,4,8 never cross the 16-lane halves (distinct M sets)
#pragma unroll
        for (int r = 0; r < 8; ++r) {
#pragma unroll
            for (int m = 8; m >= 1; m >>= 1) {
                float ov0 = __shfl_xor(bestV0[r], m);
                int   oi0 = __shfl_xor(bestI0[r], m);
                if (ov0 > bestV0[r] || (ov0 == bestV0[r] && oi0 < bestI0[r])) {
                    bestV0[r] = ov0; bestI0[r] = oi0;
                }
                float ov1 = __shfl_xor(bestV1[r], m);
                int   oi1 = __shfl_xor(bestI1[r], m);
                if (ov1 > bestV1[r] || (ov1 == bestV1[r] && oi1 < bestI1[r])) {
                    bestV1[r] = ov1; bestI1[r] = oi1;
                }
            }
        }

        // indices out: D-layout rows M = r (lanes 0-15) and M = r+8 (16-31)
        if (l16 == 0) {
#pragma unroll
            for (int r = 0; r < 8; ++r) {
                int m = r + hf * 8;
                outInd[(rowBase + waveRow + m) * NCODE + cbk]      = bestI0[r];
                outInd[(rowBase + waveRow + 16 + m) * NCODE + cbk] = bestI1[r];
            }
        }

        // ---- residual update (exact f32 codebook) + loss ----
        float lss = 0.f;
#pragma unroll
        for (int m = 0; m < 16; ++m) {
            const int idx = __shfl(bestI0[m & 7], (m < 8) ? 0 : 16);
            const float* wr = cbf32 + ((size_t)cbk * KCB + idx) * DIM;
            float* rr = myRes + m * DIM;
#pragma unroll
            for (int j = 0; j < DIM / 32; ++j) {
                float nv = rr[j * 32 + lane] - wr[j * 32 + lane];
                rr[j * 32 + lane] = nv;
                lss += nv * nv;
            }
        }
#pragma unroll
        for (int m = 0; m < 16; ++m) {
            const int idx = __shfl(bestI1[m & 7], (m < 8) ? 0 : 16);
            const float* wr = cbf32 + ((size_t)cbk * KCB + idx) * DIM;
            float* rr = myRes + (16 + m) * DIM;
#pragma unroll
            for (int j = 0; j < DIM / 32; ++j) {
                float nv = rr[j * 32 + lane] - wr[j * 32 + lane];
                rr[j * 32 + lane] = nv;
                lss += nv * nv;
            }
        }
#pragma unroll
        for (int m = 16; m >= 1; m >>= 1) lss += __shfl_xor(lss, m);
        // vq_loss + commitment_loss = 2 * mean(new_residual^2)
        if (lane == 0) atomicAdd(outLoss, lss * (2.0f / 16777216.0f));
        __syncthreads();
    }

    // ---- quant_sum = z - final_residual ----
    {
        const float4* zsrc = (const float4*)(z + rowBase * DIM);
        const float4* rsrc = (const float4*)ldsRes;
        float4* dst = (float4*)(outQ + rowBase * DIM);
        for (int i = 0; i < (ROWS_PER_BLOCK * DIM / 4) / TPB; ++i) {
            float4 zv = zsrc[tid + i * TPB];
            float4 rv = rsrc[tid + i * TPB];
            dst[tid + i * TPB] = make_float4(zv.x - rv.x, zv.y - rv.y,
                                             zv.z - rv.z, zv.w - rv.w);
        }
    }
}

extern "C" void kernel_launch(void* const* d_in, const int* in_sizes, int n_in,
                              void* d_out, int out_size, void* d_ws, size_t ws_size,
                              hipStream_t stream) {
    const float* z  = (const float*)d_in[0];                       // [65536,256]
    const float* cb = (const float*)d_in[1];                       // [4,1024,256]

    __bf16* wbf = (__bf16*)d_ws;                                   // 2 MB
    float*  w2h = (float*)((char*)d_ws + (size_t)NCODE * KCB * DIM * 2);

    float* outQ    = (float*)d_out;                                // [65536,256]
    int*   outInd  = (int*)(outQ + (size_t)N_TOK * DIM);           // [65536,4]
    float* outLoss = (float*)(outInd + (size_t)N_TOK * NCODE);     // scalar

    (void)hipFuncSetAttribute((const void*)rq_main,
                              hipFuncAttributeMaxDynamicSharedMemorySize,
                              LDS_TOTAL);

    rq_prep<<<(NCODE * KCB) / 8, 256, 0, stream>>>(cb, wbf, w2h, outLoss);
    rq_main<<<N_TOK / ROWS_PER_BLOCK, TPB, LDS_TOTAL, stream>>>(
        z, cb, wbf, w2h, outQ, outInd, outLoss);
}
